// GAT_GCN_Transformer_NN_shallow_3728031613176
// MI455X (gfx1250) — compile-verified, matrix operand order checked by reading, not run on hardware
//
#include <hip/hip_runtime.h>

// ---------------------------------------------------------------------------
// GAT_GCN_Transformer forward for MI455X (gfx1250, wave32, WMMA).
// Dense math runs through v_wmma_f32_16x16x32_f16 (f16 in, f32 acc).
// GEMM stages tiles with CDNA5 async global->LDS copies (ASYNCcnt) and
// double-buffers LDS so loads overlap WMMA. Attention is flash-style
// (no NxN materialization), single-wave blocks with wave-internal LDS
// ordering (no hardware barriers in the inner loop).
// ---------------------------------------------------------------------------

typedef __attribute__((ext_vector_type(16))) _Float16 v16h;
typedef __attribute__((ext_vector_type(8)))  _Float16 v8h;
typedef __attribute__((ext_vector_type(8)))  float    v8f;

#define GEMM_BM 64
#define GEMM_BN 128
#define GEMM_BK 32
#define LDS_AS  40   // LDS stride in halfs (80B rows -> spread banks)
#define DPMAX   800  // max padded head dim (enc2: 780 -> 800)

// ---- CDNA5 async copy: global -> LDS, tracked with ASYNCcnt ---------------
__device__ __forceinline__ void async_load_b128(void* lds, const void* gaddr)
{
    unsigned loff = (unsigned)(size_t)lds;   // low 32 bits = LDS offset
    asm volatile("global_load_async_to_lds_b128 %0, %1, off"
                 :: "v"(loff), "v"(gaddr)
                 : "memory");
}
__device__ __forceinline__ void wait_async0()
{
    asm volatile("s_wait_asynccnt 0x0" ::: "memory");
}
// compiler-ordering barrier only (intra-wave LDS is in-order on CDNA5)
__device__ __forceinline__ void wave_sync()
{
    __builtin_amdgcn_wave_barrier();
}

// ---------------------------------------------------------------------------
// Tiled WMMA GEMM:  C[M,N] = A[M,Kp] * W[N,Kp]^T (+bias)(+relu)
// A, W are f16 row-major, K padded to a multiple of 32 (zero filled).
// 256 threads = 8 waves; wave (wm,wn) owns a 16x64 strip (4 WMMA acc tiles).
// Double-buffered LDS; tiles streamed with global_load_async_to_lds_b128.
// ---------------------------------------------------------------------------
__global__ __launch_bounds__(256)
void k_gemm_f16(const _Float16* __restrict__ A, int lda,
                const _Float16* __restrict__ W, int ldw,
                float* __restrict__ C, int ldc,
                const float* __restrict__ bias,
                int M, int N, int Kp, int relu)
{
    __shared__ _Float16 As[2][GEMM_BM * LDS_AS];
    __shared__ _Float16 Bs[2][GEMM_BN * LDS_AS];

    const int t    = threadIdx.x;
    const int wid  = t >> 5;
    const int lane = t & 31;
    const int wm   = wid & 3;          // 4 wave rows  -> BM=64
    const int wn   = wid >> 2;         // 2 wave cols  -> BN=128 (64 each)
    const int lh   = lane >> 4;        // lane half
    const int lm   = lane & 15;
    const int bm   = blockIdx.x * GEMM_BM;
    const int bn   = blockIdx.y * GEMM_BN;

    // staging coords (loop-invariant; rows/cols clamped, masked at C store)
    const int ar  = t >> 2;                       // A: 64 rows x 4 chunks
    const int ac  = (t & 3) * 8;
    const int gra = (bm + ar < M) ? (bm + ar) : (M - 1);
    const int br0 = t >> 2,          bc0 = (t & 3) * 8;          // B slot t
    const int br1 = (t + 256) >> 2,  bc1 = ((t + 256) & 3) * 8;  // B slot t+256
    const int gn0 = (bn + br0 < N) ? (bn + br0) : (N - 1);
    const int gn1 = (bn + br1 < N) ? (bn + br1) : (N - 1);

    auto stage = [&](int buf, int kb) {
        async_load_b128(&As[buf][ar  * LDS_AS + ac],  A + (size_t)gra * lda + kb + ac);
        async_load_b128(&Bs[buf][br0 * LDS_AS + bc0], W + (size_t)gn0 * ldw + kb + bc0);
        async_load_b128(&Bs[buf][br1 * LDS_AS + bc1], W + (size_t)gn1 * ldw + kb + bc1);
    };

    v8f acc[4] = {};
    stage(0, 0);
    int buf = 0;

    for (int kb = 0; kb < Kp; kb += GEMM_BK) {
        wait_async0();          // my wave's async stores to LDS are done
        __syncthreads();        // all waves' stores are done
        if (kb + GEMM_BK < Kp) stage(buf ^ 1, kb + GEMM_BK);  // prefetch next

        // A fragment: lane holds row wm*16+lm; k runs lh*8..+7, lh*8+16..+23
        const _Float16* ap = &As[buf][(wm * 16 + lm) * LDS_AS + lh * 8];
        v8h a0 = *(const v8h*)ap;
        v8h a1 = *(const v8h*)(ap + 16);
        v16h a = __builtin_shufflevector(a0, a1, 0,1,2,3,4,5,6,7,8,9,10,11,12,13,14,15);

        #pragma unroll
        for (int tn = 0; tn < 4; ++tn) {
            const _Float16* bp = &Bs[buf][(wn * 64 + tn * 16 + lm) * LDS_AS + lh * 8];
            v8h b0 = *(const v8h*)bp;
            v8h b1 = *(const v8h*)(bp + 16);
            v16h b = __builtin_shufflevector(b0, b1, 0,1,2,3,4,5,6,7,8,9,10,11,12,13,14,15);
            acc[tn] = __builtin_amdgcn_wmma_f32_16x16x32_f16(
                false, a, false, b, (short)0, acc[tn], false, false);
        }
        __syncthreads();        // everyone done reading buf before overwrite
        buf ^= 1;
    }

    // epilogue: C layout = lane holds col lm, rows (lh*8 + r)
    #pragma unroll
    for (int tn = 0; tn < 4; ++tn) {
        int gcol = bn + wn * 64 + tn * 16 + lm;
        if (gcol >= N) continue;
        float bv = bias ? bias[gcol] : 0.f;
        #pragma unroll
        for (int r = 0; r < 8; ++r) {
            int grow = bm + wm * 16 + lh * 8 + r;
            if (grow < M) {
                float v = acc[tn][r] + bv;
                if (relu) v = fmaxf(v, 0.f);
                C[(size_t)grow * ldc + gcol] = v;
            }
        }
    }
}

// ---------------------------------------------------------------------------
// Flash attention (nhead=1), one wave per 16 query rows, KV tiles of 32.
// O accumulator in LDS, chunk-major [c][col][row] so C operands are b128s.
// V B-fragments load straight from global (coalesced across column lanes).
// ---------------------------------------------------------------------------
__global__ __launch_bounds__(32)
void k_attn(const float* __restrict__ Q, const float* __restrict__ K,
            const float* __restrict__ V, _Float16* __restrict__ O16,
            int N, int dp, float scale)
{
    __shared__ alignas(32) float    Os[(DPMAX / 16) * 256]; // [c][col][row]
    __shared__ alignas(16) _Float16 Ps[16 * 32];            // P tile (A-frag src)

    const int lane = threadIdx.x;
    const int lh   = lane >> 4;
    const int lm   = lane & 15;
    const int q0   = blockIdx.x * 16;
    const int nch  = dp >> 4;

    for (int i = lane; i < nch * 256; i += 32) Os[i] = 0.f;

    float mrow[8], lrow[8];
    #pragma unroll
    for (int r = 0; r < 8; ++r) { mrow[r] = -3.0e38f; lrow[r] = 0.f; }

    const float* qrow = Q + (size_t)(q0 + lm) * dp + lh * 8;

    for (int kv = 0; kv < N; kv += 32) {
        // ---- S = Q @ K_tile^T for 32 keys (two 16x16 tiles) ----
        v8f s0 = {}, s1 = {};
        const float* kr0 = K + (size_t)(kv + lm) * dp + lh * 8;
        const float* kr1 = K + (size_t)(kv + 16 + lm) * dp + lh * 8;
        for (int kc = 0; kc < dp; kc += 32) {
            v16h a, b0, b1;
            #pragma unroll
            for (int i = 0; i < 8; ++i) {
                a[i]      = (_Float16)qrow[kc + i];
                a[i + 8]  = (_Float16)qrow[kc + 16 + i];
                b0[i]     = (_Float16)kr0[kc + i];
                b0[i + 8] = (_Float16)kr0[kc + 16 + i];
                b1[i]     = (_Float16)kr1[kc + i];
                b1[i + 8] = (_Float16)kr1[kc + 16 + i];
            }
            s0 = __builtin_amdgcn_wmma_f32_16x16x32_f16(false, a, false, b0,
                                                        (short)0, s0, false, false);
            s1 = __builtin_amdgcn_wmma_f32_16x16x32_f16(false, a, false, b1,
                                                        (short)0, s1, false, false);
        }

        // ---- online softmax: row stats over 32 cols ----
        float alpha[8];
        #pragma unroll
        for (int r = 0; r < 8; ++r) {
            float v0 = s0[r] * scale, v1 = s1[r] * scale;
            float mx = fmaxf(v0, v1);
            #pragma unroll
            for (int off = 1; off < 16; off <<= 1)
                mx = fmaxf(mx, __shfl_xor(mx, off, 32));
            float mnew = fmaxf(mrow[r], mx);
            alpha[r]   = __expf(mrow[r] - mnew);
            float p0 = __expf(v0 - mnew);
            float p1 = __expf(v1 - mnew);
            float rs = p0 + p1;
            #pragma unroll
            for (int off = 1; off < 16; off <<= 1)
                rs += __shfl_xor(rs, off, 32);
            lrow[r] = lrow[r] * alpha[r] + rs;
            mrow[r] = mnew;
            Ps[(lh * 8 + r) * 32 + lm]      = (_Float16)p0;
            Ps[(lh * 8 + r) * 32 + 16 + lm] = (_Float16)p1;
        }
        wave_sync();   // intra-wave LDS ordering only; no hardware barrier

        // ---- P as 16x32 A-fragment ----
        v16h pa;
        {
            const _Float16* p = &Ps[lm * 32 + lh * 8];
            #pragma unroll
            for (int i = 0; i < 8; ++i) { pa[i] = p[i]; pa[i + 8] = p[16 + i]; }
        }
        wave_sync();

        // ---- O = O*alpha + P @ V_tile, 16 output cols per WMMA ----
        for (int c = 0; c < nch; ++c) {
            // V B-fragment straight from global: col = lm, k = lh*8+i (+16)
            const float* vp = V + (size_t)kv * dp + c * 16 + lm;
            v16h vb;
            #pragma unroll
            for (int i = 0; i < 8; ++i) {
                vb[i]     = (_Float16)vp[(size_t)(lh * 8 + i) * dp];
                vb[i + 8] = (_Float16)vp[(size_t)(lh * 8 + 16 + i) * dp];
            }
            float* ob = &Os[c * 256 + lm * 16 + lh * 8];
            v8f o = *(const v8f*)ob;
            #pragma unroll
            for (int r = 0; r < 8; ++r) o[r] *= alpha[r];
            o = __builtin_amdgcn_wmma_f32_16x16x32_f16(false, pa, false, vb,
                                                       (short)0, o, false, false);
            *(v8f*)ob = o;
        }
        wave_sync();
    }

    // ---- epilogue: O/l -> f16 padded output ----
    for (int c = 0; c < nch; ++c) {
        #pragma unroll
        for (int r = 0; r < 8; ++r) {
            float inv = 1.f / lrow[r];
            O16[(size_t)(q0 + lh * 8 + r) * dp + c * 16 + lm] =
                (_Float16)(Os[c * 256 + lm * 16 + lh * 8 + r] * inv);
        }
    }
}

// ---------------------------------------------------------------------------
// Elementwise / scatter helpers
// ---------------------------------------------------------------------------
__global__ void k_cvt(const float* __restrict__ src, _Float16* __restrict__ dst,
                      int M, int K, int Kp)
{
    int i = blockIdx.x * blockDim.x + threadIdx.x;
    if (i >= M * Kp) return;
    int m = i / Kp, c = i - m * Kp;
    dst[i] = (c < K) ? (_Float16)src[(size_t)m * K + c] : (_Float16)0.f;
}

__global__ void k_splitpad(const float* __restrict__ qkv, float* __restrict__ dst,
                           int N, int d, int dp)
{
    int i = blockIdx.x * blockDim.x + threadIdx.x;
    int total = N * dp;
    if (i >= 3 * total) return;
    int which = i / total;
    int rem   = i - which * total;
    int n = rem / dp, c = rem - n * dp;
    dst[i] = (c < d) ? qkv[(size_t)n * 3 * d + which * d + c] : 0.f;
}

__global__ __launch_bounds__(32)
void k_add_ln(const float* __restrict__ x, const float* __restrict__ a,
              const float* __restrict__ g, const float* __restrict__ b,
              float* __restrict__ y, int d)
{
    int row = blockIdx.x;
    const float* xr = x + (size_t)row * d;
    const float* ar = a + (size_t)row * d;
    float s = 0.f, ss = 0.f;
    for (int i = threadIdx.x; i < d; i += 32) {
        float v = xr[i] + ar[i]; s += v; ss += v * v;
    }
    #pragma unroll
    for (int off = 16; off; off >>= 1) {
        s  += __shfl_xor(s,  off, 32);
        ss += __shfl_xor(ss, off, 32);
    }
    float mean = s / d;
    float var  = ss / d - mean * mean;
    float inv  = rsqrtf(var + 1e-5f);
    float* yr = y + (size_t)row * d;
    for (int i = threadIdx.x; i < d; i += 32)
        yr[i] = ((xr[i] + ar[i]) - mean) * inv * g[i] + b[i];
}

__global__ void k_fill(float* p, float v, int n)
{
    int i = blockIdx.x * blockDim.x + threadIdx.x;
    if (i < n) p[i] = v;
}

__global__ void k_bias_act(float* p, const float* __restrict__ b, int M, int Nc, int relu)
{
    int i = blockIdx.x * blockDim.x + threadIdx.x;
    if (i >= M * Nc) return;
    int c = i % Nc;
    float v = p[i] + b[c];
    if (relu) v = fmaxf(v, 0.f);
    p[i] = v;
}

__device__ __forceinline__ void atomicMaxF(float* addr, float val)
{
    int iv = __float_as_int(val);
    if (iv >= 0) atomicMax((int*)addr, iv);
    else         atomicMin((unsigned int*)addr, (unsigned int)iv);
}

__device__ __forceinline__ float leaky02(float v) { return v > 0.f ? v : 0.2f * v; }

__global__ void k_gat_scores(const float* __restrict__ xp,
                             const float* __restrict__ asrc,
                             const float* __restrict__ adst,
                             float* __restrict__ as_, float* __restrict__ ad_, int Nn)
{
    int i = blockIdx.x * blockDim.x + threadIdx.x;
    if (i >= Nn * 10) return;
    int n = i / 10, h = i - n * 10;
    const float* xr = xp + (size_t)n * 780 + h * 78;
    const float* sw = asrc + h * 78;
    const float* dw = adst + h * 78;
    float vs = 0.f, vd = 0.f;
    for (int j = 0; j < 78; ++j) { float v = xr[j]; vs += v * sw[j]; vd += v * dw[j]; }
    as_[i] = vs; ad_[i] = vd;
}

__global__ void k_gat_e1(const int* __restrict__ ei, const float* __restrict__ as_,
                         const float* __restrict__ ad_, float* gm, int E, int Nn)
{
    int e = blockIdx.x * blockDim.x + threadIdx.x;
    if (e >= E + Nn) return;
    int s = (e < E) ? ei[e] : (e - E);
    int d = (e < E) ? ei[E + e] : (e - E);
    for (int h = 0; h < 10; ++h)
        atomicMaxF(&gm[d * 10 + h], leaky02(as_[s * 10 + h] + ad_[d * 10 + h]));
}

__global__ void k_gat_e2(const int* __restrict__ ei, const float* __restrict__ as_,
                         const float* __restrict__ ad_, const float* __restrict__ gm,
                         float* gs, int E, int Nn)
{
    int e = blockIdx.x * blockDim.x + threadIdx.x;
    if (e >= E + Nn) return;
    int s = (e < E) ? ei[e] : (e - E);
    int d = (e < E) ? ei[E + e] : (e - E);
    for (int h = 0; h < 10; ++h) {
        float v = leaky02(as_[s * 10 + h] + ad_[d * 10 + h]);
        atomicAdd(&gs[d * 10 + h], __expf(v - gm[d * 10 + h]));
    }
}

__global__ void k_gat_e3(const int* __restrict__ ei, const float* __restrict__ as_,
                         const float* __restrict__ ad_, const float* __restrict__ gm,
                         const float* __restrict__ gs, const float* __restrict__ xp,
                         float* out, int E, int Nn)
{
    int e = blockIdx.x * blockDim.x + threadIdx.x;
    if (e >= E + Nn) return;
    int s = (e < E) ? ei[e] : (e - E);
    int d = (e < E) ? ei[E + e] : (e - E);
    for (int h = 0; h < 10; ++h) {
        float v = leaky02(as_[s * 10 + h] + ad_[d * 10 + h]);
        float alpha = __expf(v - gm[d * 10 + h]) / (gs[d * 10 + h] + 1e-16f);
        const float* xs = xp + (size_t)s * 780 + h * 78;
        float*       od = out + (size_t)d * 780 + h * 78;
        for (int j = 0; j < 78; ++j) atomicAdd(&od[j], alpha * xs[j]);
    }
}

__global__ void k_deg(const int* __restrict__ ei, float* deg, int E, int Nn)
{
    int e = blockIdx.x * blockDim.x + threadIdx.x;
    if (e >= E + Nn) return;
    int d = (e < E) ? ei[E + e] : (e - E);
    atomicAdd(&deg[d], 1.f);
}

__global__ void k_dinv(float* deg, int Nn)
{
    int i = blockIdx.x * blockDim.x + threadIdx.x;
    if (i < Nn) deg[i] = rsqrtf(fmaxf(deg[i], 1.f));
}

__global__ void k_gcn_agg(const int* __restrict__ ei, const float* __restrict__ dinv,
                          const float* __restrict__ xw, float* out, int E, int Nn)
{
    int e = blockIdx.x * blockDim.x + threadIdx.x;
    if (e >= E + Nn) return;
    int s = (e < E) ? ei[e] : (e - E);
    int d = (e < E) ? ei[E + e] : (e - E);
    float w = dinv[s] * dinv[d];
    const float* xs = xw + (size_t)s * 780;
    float*       od = out + (size_t)d * 780;
    for (int j = 0; j < 780; ++j) atomicAdd(&od[j], w * xs[j]);
}

__global__ void k_pool(const float* __restrict__ h, const int* __restrict__ batch,
                       float* xmax, float* xsum, float* cnt, int Nn)
{
    int n = blockIdx.x * blockDim.x + threadIdx.x;
    if (n >= Nn) return;
    int g = batch[n];
    atomicAdd(&cnt[g], 1.f);
    const float* hr = h + (size_t)n * 780;
    for (int j = 0; j < 780; ++j) {
        atomicMaxF(&xmax[g * 780 + j], hr[j]);
        atomicAdd(&xsum[g * 780 + j], hr[j]);
    }
}

__global__ void k_pool_fin(const float* __restrict__ xmax, const float* __restrict__ xsum,
                           const float* __restrict__ cnt, float* xg, int G)
{
    int i = blockIdx.x * blockDim.x + threadIdx.x;
    if (i >= G * 1560) return;
    int g = i / 1560, j = i - g * 1560;
    xg[i] = (j < 780) ? xmax[g * 780 + j]
                      : xsum[g * 780 + (j - 780)] / fmaxf(cnt[g], 1.f);
}

__global__ void k_concat_cvt(const float* __restrict__ a, const float* __restrict__ b,
                             _Float16* dst, int M, int Ka, int Kb)
{
    int i = blockIdx.x * blockDim.x + threadIdx.x;
    int Kt = Ka + Kb;
    if (i >= M * Kt) return;
    int m = i / Kt, c = i - m * Kt;
    dst[i] = (c < Ka) ? (_Float16)a[(size_t)m * Ka + c]
                      : (_Float16)b[(size_t)m * Kb + (c - Ka)];
}

__global__ void k_final(const float* __restrict__ c2, const float* __restrict__ ow,
                        const float* __restrict__ ob, float* out, int G)
{
    int g = blockIdx.x * blockDim.x + threadIdx.x;
    if (g >= G) return;
    float s = ob[0];
    for (int j = 0; j < 128; ++j) s += c2[g * 128 + j] * ow[j];
    out[g] = 1.f / (1.f + __expf(-s));
}

// ---------------------------------------------------------------------------
// Host orchestration
// ---------------------------------------------------------------------------
static inline int ceil_div(int a, int b) { return (a + b - 1) / b; }
static inline int pad32(int k) { return (k + 31) & ~31; }

extern "C" void kernel_launch(void* const* d_in, const int* in_sizes, int n_in,
                              void* d_out, int out_size, void* d_ws, size_t ws_size,
                              hipStream_t stream)
{
    (void)in_sizes; (void)n_in; (void)out_size; (void)ws_size;

    const int N = 8000, E = 256000, G = 64;
    const int D1 = 78, D2 = 780, FF = 2048;
    const int Etot = E + N;

    const float* x      = (const float*)d_in[0];
    const int*   ei     = (const int*)  d_in[1];
    const int*   batch  = (const int*)  d_in[2];
    const float* target = (const float*)d_in[3];
    const float* e1_in_w = (const float*)d_in[4];  const float* e1_in_b = (const float*)d_in[5];
    const float* e1_ow   = (const float*)d_in[6];  const float* e1_ob   = (const float*)d_in[7];
    const float* e1_l1w  = (const float*)d_in[8];  const float* e1_l1b  = (const float*)d_in[9];
    const float* e1_l2w  = (const float*)d_in[10]; const float* e1_l2b  = (const float*)d_in[11];
    const float* e1_g1   = (const float*)d_in[12]; const float* e1_b1   = (const float*)d_in[13];
    const float* e1_g2   = (const float*)d_in[14]; const float* e1_b2   = (const float*)d_in[15];
    const float* e2_in_w = (const float*)d_in[16]; const float* e2_in_b = (const float*)d_in[17];
    const float* e2_ow   = (const float*)d_in[18]; const float* e2_ob   = (const float*)d_in[19];
    const float* e2_l1w  = (const float*)d_in[20]; const float* e2_l1b  = (const float*)d_in[21];
    const float* e2_l2w  = (const float*)d_in[22]; const float* e2_l2b  = (const float*)d_in[23];
    const float* e2_g1   = (const float*)d_in[24]; const float* e2_b1   = (const float*)d_in[25];
    const float* e2_g2   = (const float*)d_in[26]; const float* e2_b2   = (const float*)d_in[27];
    const float* gat_w   = (const float*)d_in[28];
    const float* att_src = (const float*)d_in[29];
    const float* att_dst = (const float*)d_in[30];
    const float* gat_b   = (const float*)d_in[31];
    const float* gcn_w   = (const float*)d_in[32]; const float* gcn_b = (const float*)d_in[33];
    const float* fcg1_w = (const float*)d_in[34]; const float* fcg1_b = (const float*)d_in[35];
    const float* fcg2_w = (const float*)d_in[36]; const float* fcg2_b = (const float*)d_in[37];
    const float* lin1_w = (const float*)d_in[38]; const float* lin1_b = (const float*)d_in[39];
    const float* lin2_w = (const float*)d_in[40]; const float* lin2_b = (const float*)d_in[41];
    const float* fc1_w  = (const float*)d_in[42]; const float* fc1_b  = (const float*)d_in[43];
    const float* fc2_w  = (const float*)d_in[44]; const float* fc2_b  = (const float*)d_in[45];
    const float* out_w  = (const float*)d_in[46]; const float* out_b  = (const float*)d_in[47];

    char* ws = (char*)d_ws;
    size_t off = 0;
    auto alloc = [&](size_t bytes) -> void* {
        off = (off + 255) & ~(size_t)255;
        void* p = ws + off;
        off += bytes;
        return p;
    };

    _Float16* W16  = (_Float16*)alloc((size_t)2400000 * 2);
    _Float16* A16  = (_Float16*)alloc((size_t)N * 2048 * 2);
    float*    QKVP = (float*)   alloc((size_t)3 * N * DPMAX * 4);
    _Float16* ATT16= (_Float16*)alloc((size_t)N * DPMAX * 2);
    float*    F1   = (float*)   alloc((size_t)N * 2048 * 4);
    float*    F2   = (float*)   alloc((size_t)N * 2340 * 4);
    float*    F3   = (float*)   alloc((size_t)N * 780 * 4);
    float*    F4   = (float*)   alloc((size_t)N * 780 * 4);
    float*    F5   = (float*)   alloc((size_t)N * 780 * 4);
    float*    as_  = (float*)   alloc((size_t)N * 10 * 4);
    float*    ad_  = (float*)   alloc((size_t)N * 10 * 4);
    float*    gm   = (float*)   alloc((size_t)N * 10 * 4);
    float*    gs   = (float*)   alloc((size_t)N * 10 * 4);
    float*    deg  = (float*)   alloc((size_t)N * 4);
    float*    xmax = (float*)   alloc((size_t)G * 780 * 4);
    float*    xsum = (float*)   alloc((size_t)G * 780 * 4);
    float*    cnt  = (float*)   alloc((size_t)G * 4);
    float*    XG0  = (float*)   alloc((size_t)G * 1560 * 4);
    float*    G1b  = (float*)   alloc((size_t)G * 1500 * 4);
    float*    T1   = (float*)   alloc((size_t)G * 256 * 4);
    float*    T2   = (float*)   alloc((size_t)G * 128 * 4);
    float*    C1   = (float*)   alloc((size_t)G * 1024 * 4);
    float*    C2   = (float*)   alloc((size_t)G * 128 * 4);

    float* xg_out  = (float*)d_out + 64;
    float* out_vec = (float*)d_out;

    auto cvt = [&](const float* src, _Float16* dst, int M, int K, int Kp) {
        int tot = M * Kp;
        k_cvt<<<ceil_div(tot, 256), 256, 0, stream>>>(src, dst, M, K, Kp);
    };
    auto gemm_raw = [&](const _Float16* A, int lda, const _Float16* Wp, int ldw,
                        float* C, int ldc, const float* bias, int M, int Nn, int Kp, int relu) {
        dim3 g(ceil_div(M, GEMM_BM), ceil_div(Nn, GEMM_BN));
        k_gemm_f16<<<g, 256, 0, stream>>>(A, lda, Wp, ldw, C, ldc, bias, M, Nn, Kp, relu);
    };
    auto gemm = [&](const float* Af, const float* Wf, float* C, int ldc,
                    const float* bias, int M, int Nn, int K, int relu) {
        int Kp = pad32(K);
        cvt(Af, A16, M, K, Kp);
        cvt(Wf, W16, Nn, K, Kp);
        gemm_raw(A16, Kp, W16, Kp, C, ldc, bias, M, Nn, Kp, relu);
    };
    auto gemmA = [&](const _Float16* A, int lda, const float* Wf, float* C, int ldc,
                     const float* bias, int M, int Nn, int K, int relu) {
        int Kp = pad32(K);
        cvt(Wf, W16, Nn, K, Kp);
        gemm_raw(A, lda, W16, Kp, C, ldc, bias, M, Nn, Kp, relu);
    };
    auto fill = [&](float* p, float v, int n) {
        k_fill<<<ceil_div(n, 256), 256, 0, stream>>>(p, v, n);
    };

    // =========================== encoder 1 (d=78) ===========================
    {
        int d = D1, dp = pad32(D1);
        gemm(x, e1_in_w, F1, 3 * d, e1_in_b, N, 3 * d, d, 0);
        int tot = 3 * N * dp;
        k_splitpad<<<ceil_div(tot, 256), 256, 0, stream>>>(F1, QKVP, N, d, dp);
        k_attn<<<N / 16, 32, 0, stream>>>(QKVP, QKVP + (size_t)N * dp,
                                          QKVP + (size_t)2 * N * dp, ATT16,
                                          N, dp, rsqrtf((float)d));
        gemmA(ATT16, dp, e1_ow, F2, d, e1_ob, N, d, d, 0);
        k_add_ln<<<N, 32, 0, stream>>>(x, F2, e1_g1, e1_b1, F3, d);
        gemm(F3, e1_l1w, F2, FF, e1_l1b, N, FF, d, 1);
        gemm(F2, e1_l2w, F1, d, e1_l2b, N, d, FF, 0);
        k_add_ln<<<N, 32, 0, stream>>>(F3, F1, e1_g2, e1_b2, F4, d);
    }

    // ================================ GAT ===================================
    {
        gemm(F4, gat_w, F5, D2, nullptr, N, D2, D1, 0);
        k_gat_scores<<<ceil_div(N * 10, 256), 256, 0, stream>>>(F5, att_src, att_dst,
                                                                as_, ad_, N);
        fill(gm, -3.0e38f, N * 10);
        fill(gs, 0.f, N * 10);
        int gE = ceil_div(Etot, 256);
        k_gat_e1<<<gE, 256, 0, stream>>>(ei, as_, ad_, gm, E, N);
        k_gat_e2<<<gE, 256, 0, stream>>>(ei, as_, ad_, gm, gs, E, N);
        fill(F1, 0.f, N * D2);
        k_gat_e3<<<gE, 256, 0, stream>>>(ei, as_, ad_, gm, gs, F5, F1, E, N);
        k_bias_act<<<ceil_div(N * D2, 256), 256, 0, stream>>>(F1, gat_b, N, D2, 1);
    }

    // =========================== encoder 2 (d=780) ==========================
    {
        int d = D2, dp = pad32(D2);
        gemm(F1, e2_in_w, F2, 3 * d, e2_in_b, N, 3 * d, d, 0);
        int tot = 3 * N * dp;
        k_splitpad<<<ceil_div(tot, 256), 256, 0, stream>>>(F2, QKVP, N, d, dp);
        k_attn<<<N / 16, 32, 0, stream>>>(QKVP, QKVP + (size_t)N * dp,
                                          QKVP + (size_t)2 * N * dp, ATT16,
                                          N, dp, rsqrtf((float)d));
        gemmA(ATT16, dp, e2_ow, F3, d, e2_ob, N, d, d, 0);
        k_add_ln<<<N, 32, 0, stream>>>(F1, F3, e2_g1, e2_b1, F4, d);
        gemm(F4, e2_l1w, F2, FF, e2_l1b, N, FF, d, 1);
        gemm(F2, e2_l2w, F3, d, e2_l2b, N, d, FF, 0);
        k_add_ln<<<N, 32, 0, stream>>>(F4, F3, e2_g2, e2_b2, F5, d);
    }

    // ================================ GCN ===================================
    {
        fill(deg, 0.f, N);
        int gE = ceil_div(Etot, 256);
        k_deg<<<gE, 256, 0, stream>>>(ei, deg, E, N);
        k_dinv<<<ceil_div(N, 256), 256, 0, stream>>>(deg, N);
        gemm(F5, gcn_w, F3, D2, nullptr, N, D2, D2, 0);
        fill(F4, 0.f, N * D2);
        k_gcn_agg<<<gE, 256, 0, stream>>>(ei, deg, F3, F4, E, N);
        k_bias_act<<<ceil_div(N * D2, 256), 256, 0, stream>>>(F4, gcn_b, N, D2, 1);
    }

    // ======================== pooling + output heads ========================
    fill(xmax, -3.0e38f, G * 780);
    fill(xsum, 0.f, G * 780);
    fill(cnt, 0.f, G);
    k_pool<<<ceil_div(N, 256), 256, 0, stream>>>(F4, batch, xmax, xsum, cnt, N);
    k_pool_fin<<<ceil_div(G * 1560, 256), 256, 0, stream>>>(xmax, xsum, cnt, XG0, G);

    gemm(XG0, fcg1_w, G1b, 1500, fcg1_b, G, 1500, 1560, 1);
    gemm(G1b, fcg2_w, xg_out, 128, fcg2_b, G, 128, 1500, 0);
    gemm(target, lin1_w, T1, 256, lin1_b, G, 256, 735, 0);
    gemm(T1, lin2_w, T2, 128, lin2_b, G, 128, 256, 0);

    k_concat_cvt<<<ceil_div(G * 256, 256), 256, 0, stream>>>(xg_out, T2, A16, G, 128, 128);
    cvt(fc1_w, W16, 1024, 256, 256);
    gemm_raw(A16, 256, W16, 256, C1, 1024, fc1_b, G, 1024, 256, 1);
    gemm(C1, fc2_w, C2, 128, fc2_b, G, 128, 1024, 1);
    k_final<<<ceil_div(G, 64), 64, 0, stream>>>(C2, out_w, out_b, out_vec, G);
}
